// FakeSetTopoLayer_76407468196371
// MI455X (gfx1250) — matrix operand order repaired and checked by reading, not compiled, not run on hardware
//
#include <hip/hip_runtime.h>

#define NN 524288
#define FF 128
#define HH 128
#define NFILT 8
#define GG 2048

typedef __attribute__((ext_vector_type(16))) _Float16 v16h;
typedef __attribute__((ext_vector_type(8)))  _Float16 v8h;
typedef __attribute__((ext_vector_type(8)))  float    v8f;
typedef __attribute__((ext_vector_type(4)))  unsigned int u32x4;
typedef __attribute__((ext_vector_type(8)))  int          i32x8;
typedef __attribute__((ext_vector_type(4)))  int          i32x4;

// ---------------------------------------------------------------------------
// WMMA fragment helpers (CDNA5 wave32 layouts, CDNA5 ISA §7.12.2)
// ---------------------------------------------------------------------------

// A-matrix 16x32 f16: lane L holds row M=L&15; lanes 0-15 K groups {0..7,16..23},
// lanes 16-31 K groups {8..15,24..31}; 2 halves per VGPR.
__device__ __forceinline__ v16h load_a_frag(const _Float16* rowbase, int ld, int kbase) {
  int lane = threadIdx.x & 31;
  int m = lane & 15, hh = lane >> 4;
  const _Float16* row = rowbase + m * ld + kbase + hh * 8;
  union { v16h v; unsigned u[8]; } r;
#pragma unroll
  for (int i = 0; i < 4; ++i) {
    r.u[i]     = *(const unsigned*)(row + 2 * i);
    r.u[i + 4] = *(const unsigned*)(row + 16 + 2 * i);
  }
  return r.v;
}

// B-matrix 32x16 f16 from an N-major (transposed) weight: lane L holds column
// N=nbase+(L&15); lanes 0-15 K groups {0..7,16..23}, lanes 16-31 K groups {8..15,24..31}.
__device__ __forceinline__ v16h load_b_frag(const _Float16* wt, int ldk, int nbase, int kbase) {
  int lane = threadIdx.x & 31;
  const _Float16* p = wt + (nbase + (lane & 15)) * ldk + kbase + (lane >> 4) * 8;
  union { v16h v; v8h h[2]; } r;
  r.h[0] = *(const v8h*)(p);
  r.h[1] = *(const v8h*)(p + 16);
  return r.v;
}

__device__ __forceinline__ v8f wmma_f32(v16h a, v16h b, v8f c) {
  return __builtin_amdgcn_wmma_f32_16x16x32_f16(
      /*neg_a=*/false, a, /*neg_b=*/false, b,
      /*c_mod=*/(short)0, c, /*reuse_a=*/false, /*reuse_b=*/false);
}

// ---------------------------------------------------------------------------
// Utility kernels
// ---------------------------------------------------------------------------

__global__ void k_zero(float* p, int n) {
  int i = blockIdx.x * 256 + threadIdx.x;
  if (i < n) p[i] = 0.0f;
}

// f32 [K][Nsrc] -> f16 N-major [Npad][ldk], zero padded
__global__ void k_prep_w(const float* __restrict__ src, _Float16* __restrict__ dst,
                         int K, int Nsrc, int Npad, int ldk) {
  int i = blockIdx.x * 256 + threadIdx.x;
  if (i >= Npad * ldk) return;
  int n = i / ldk, k = i - n * ldk;
  float v = (n < Nsrc && k < K) ? src[k * Nsrc + n] : 0.0f;
  dst[i] = (_Float16)v;
}

// x f32 [N,128] -> x0 f16 [N,160] (8 halves / thread, 16B stores);
// cols 128..159 zeroed (filt fills 128..135)
__global__ void k_convert_x(const float* __restrict__ x, _Float16* __restrict__ x0) {
  int i = blockIdx.x * 256 + threadIdx.x;         // NN*20 chunks of 8 halves
  if (i >= NN * 20) return;
  int row = i / 20, c8 = (i - row * 20) * 8;
  v8h o = {};
  if (c8 < FF) {
    const float4* xp = (const float4*)(x + row * FF + c8);
    float4 a = xp[0], b = xp[1];
    o[0] = (_Float16)a.x; o[1] = (_Float16)a.y; o[2] = (_Float16)a.z; o[3] = (_Float16)a.w;
    o[4] = (_Float16)b.x; o[5] = (_Float16)b.y; o[6] = (_Float16)b.z; o[7] = (_Float16)b.w;
  }
  *(v8h*)(x0 + (long)row * 160 + c8) = o;
}

// ---------------------------------------------------------------------------
// Filtration MLP: filt = relu(x@W1+b1)@W2+b2, written into x0[:,128:136]
// ---------------------------------------------------------------------------
__global__ __launch_bounds__(256) void k_filt(const _Float16* __restrict__ w1t,
                                              const _Float16* __restrict__ w2t,
                                              const float* __restrict__ b1,
                                              const float* __restrict__ b2,
                                              _Float16* __restrict__ x0) {
  __shared__ _Float16 w1l[128 * 128];   // reused as hidden staging after GEMM1
  __shared__ _Float16 w2l[16 * 128];
  {
    const uint4* s1 = (const uint4*)w1t; uint4* d1 = (uint4*)w1l;
    for (int i = threadIdx.x; i < (128 * 128) / 8; i += 256) d1[i] = s1[i];
    const uint4* s2 = (const uint4*)w2t; uint4* d2 = (uint4*)w2l;
    for (int i = threadIdx.x; i < (16 * 128) / 8; i += 256) d2[i] = s2[i];
  }
  __syncthreads();

  int wave = threadIdx.x >> 5, lane = threadIdx.x & 31;
  int rowbase = blockIdx.x * 128 + wave * 16;
  const _Float16* arow = x0 + (long)rowbase * 160;
  __builtin_prefetch(arow, 0, 0);

  v8f acc[8] = {};
#pragma unroll
  for (int ks = 0; ks < 4; ++ks) {
    v16h a = load_a_frag(arow, 160, ks * 32);
    v16h b[8];
#pragma unroll
    for (int t = 0; t < 8; ++t) b[t] = load_b_frag(w1l, 128, t * 16, ks * 32);
#pragma unroll
    for (int t = 0; t < 8; ++t) acc[t] = wmma_f32(a, b[t], acc[t]);
  }

  __syncthreads();                       // everyone done reading w1l
  _Float16* hl = w1l + wave * (16 * 128);
#pragma unroll
  for (int t = 0; t < 8; ++t) {
    int n = t * 16 + (lane & 15);
    float bias = b1[n];
#pragma unroll
    for (int r = 0; r < 8; ++r) {
      int m = r + 8 * (lane >> 4);
      hl[m * 128 + n] = (_Float16)fmaxf(acc[t][r] + bias, 0.0f);
    }
  }
  __syncthreads();                       // hidden tiles visible

  v8f acc2 = {};
#pragma unroll
  for (int ks = 0; ks < 4; ++ks) {
    v16h a = load_a_frag(hl, 128, ks * 32);
    v16h b = load_b_frag(w2l, 128, 0, ks * 32);
    acc2 = wmma_f32(a, b, acc2);
  }
  int n = lane & 15;
  if (n < NFILT) {
    float bias = b2[n];
#pragma unroll
    for (int r = 0; r < 8; ++r) {
      int m = r + 8 * (lane >> 4);
      x0[(long)(rowbase + m) * 160 + 128 + n] = (_Float16)(acc2[r] + bias);
    }
  }
}

// ---------------------------------------------------------------------------
// Scatter-sum of x0 (136 live cols) into per-graph sums + counts
// ---------------------------------------------------------------------------
__global__ void k_scatter1(const _Float16* __restrict__ x0, const int* __restrict__ batch,
                           float* __restrict__ sum1, float* __restrict__ cnt) {
  int i = blockIdx.x * 256 + threadIdx.x;
  if (i >= NN * 136) return;
  int row = i / 136, col = i - row * 136;
  int g = batch[row];
  float v = (float)x0[(long)row * 160 + col];
  atomicAdd(&sum1[g * 160 + col], v);
  if (col == 0) atomicAdd(&cnt[g], 1.0f);
}

// ---------------------------------------------------------------------------
// corr = (sum/cnt) @ Lwt  for G graphs (A built in LDS as f16)
// ---------------------------------------------------------------------------
__global__ __launch_bounds__(256) void k_meancorr(const float* __restrict__ sum,
                                                  const float* __restrict__ cnt,
                                                  const _Float16* __restrict__ lwt,
                                                  float* __restrict__ corr,
                                                  int ldsum, int ksteps) {
  __shared__ _Float16 xml[128 * 160];
  int gbase = blockIdx.x * 128;
  int tot = 128 * ldsum;
  for (int i = threadIdx.x; i < tot; i += 256) {
    int r = i / ldsum, c = i - r * ldsum;
    float cc = cnt[gbase + r];
    xml[i] = (_Float16)(sum[(gbase + r) * ldsum + c] / fmaxf(cc, 1.0f));
  }
  __syncthreads();

  int wave = threadIdx.x >> 5, lane = threadIdx.x & 31;
  const _Float16* arow = xml + wave * 16 * ldsum;
  v8f acc[8] = {};
  for (int ks = 0; ks < ksteps; ++ks) {
    v16h a = load_a_frag(arow, ldsum, ks * 32);
    v16h b[8];
#pragma unroll
    for (int t = 0; t < 8; ++t) b[t] = load_b_frag(lwt, ldsum, t * 16, ks * 32);
#pragma unroll
    for (int t = 0; t < 8; ++t) acc[t] = wmma_f32(a, b[t], acc[t]);
  }
#pragma unroll
  for (int t = 0; t < 8; ++t) {
    int n = t * 16 + (lane & 15);
#pragma unroll
    for (int r = 0; r < 8; ++r) {
      int m = r + 8 * (lane >> 4);
      corr[(gbase + wave * 16 + m) * 128 + n] = acc[t][r];
    }
  }
}

// ---------------------------------------------------------------------------
// Layer 1: h = relu(x0@G1 + b - corr1[batch]); also scatter-sum h per graph
// ---------------------------------------------------------------------------
__global__ __launch_bounds__(256) void k_layer1(const _Float16* __restrict__ x0,
                                                const _Float16* __restrict__ g1wt,
                                                const float* __restrict__ g1b,
                                                const float* __restrict__ corr1,
                                                const int* __restrict__ batch,
                                                _Float16* __restrict__ h,
                                                float* __restrict__ sum2) {
  __shared__ _Float16 wl[128 * 160];
  {
    const uint4* s = (const uint4*)g1wt; uint4* d = (uint4*)wl;
    for (int i = threadIdx.x; i < (128 * 160) / 8; i += 256) d[i] = s[i];
  }
  __syncthreads();

  int wave = threadIdx.x >> 5, lane = threadIdx.x & 31;
  int rowbase = blockIdx.x * 128 + wave * 16;
  const _Float16* arow = x0 + (long)rowbase * 160;
  __builtin_prefetch(arow, 0, 0);

  v8f acc[8] = {};
#pragma unroll
  for (int ks = 0; ks < 5; ++ks) {
    v16h a = load_a_frag(arow, 160, ks * 32);
    v16h b[8];
#pragma unroll
    for (int t = 0; t < 8; ++t) b[t] = load_b_frag(wl, 160, t * 16, ks * 32);
#pragma unroll
    for (int t = 0; t < 8; ++t) acc[t] = wmma_f32(a, b[t], acc[t]);
  }

  int hh = lane >> 4;
  int gg[8];
#pragma unroll
  for (int r = 0; r < 8; ++r) gg[r] = batch[rowbase + hh * 8 + r];
#pragma unroll
  for (int t = 0; t < 8; ++t) {
    int n = t * 16 + (lane & 15);
    float bias = g1b[n];
#pragma unroll
    for (int r = 0; r < 8; ++r) {
      int row = rowbase + hh * 8 + r;
      float v = fmaxf(acc[t][r] + bias - corr1[gg[r] * 128 + n], 0.0f);
      h[(long)row * 128 + n] = (_Float16)v;
      atomicAdd(&sum2[gg[r] * 128 + n], v);
    }
  }
}

// ---------------------------------------------------------------------------
// Layer 2: y = h@G2 + b - corr2[batch]; fused batchnorm partial sums (LDS).
// Weight staged into LDS with the Tensor Data Mover (one D# issued by wave 0).
// ---------------------------------------------------------------------------
__global__ __launch_bounds__(256) void k_layer2(const _Float16* __restrict__ h,
                                                const _Float16* __restrict__ g2wt,
                                                const float* __restrict__ g2b,
                                                const float* __restrict__ corr2,
                                                const int* __restrict__ batch,
                                                float* __restrict__ y,
                                                float* __restrict__ bn_sum,
                                                float* __restrict__ bn_sq) {
  __shared__ _Float16 wl[128 * 128];    // 32 KB weight tile
  __shared__ float bsum[128], bsq[128];

#if __has_builtin(__builtin_amdgcn_tensor_load_to_lds)
  if (threadIdx.x < 32) {               // one wave issues the TDM descriptor
    unsigned lds_base = (unsigned)(size_t)(&wl[0]);
    unsigned long long ga = (unsigned long long)(size_t)g2wt;
    const unsigned elems = (128u * 128u * 2u) / 8u;   // 4096 x 8-byte elements
    // D# group 0: count=1 (valid user descriptor), lds_addr, global_addr, type=2
    u32x4 g0;
    g0[0] = 1u;
    g0[1] = lds_base;
    g0[2] = (unsigned)(ga & 0xffffffffu);
    g0[3] = (unsigned)((ga >> 32) & 0x01ffffffu) | (2u << 30);
    // D# group 1: data_size=3 (8B); tensor_dim0=tile_dim0=4096; dims1=1; stride0=4096
    i32x8 g1;
    g1[0] = (int)(3u << 16);            // workgroup_mask=0, data_size=8B
    g1[1] = (int)(elems << 16);         // tensor_dim0[15:0] in bits 31:16
    g1[2] = (int)(1u << 16);            // tensor_dim0 hi=0, tensor_dim1[15:0]=1
    g1[3] = (int)(elems << 16);         // tensor_dim1 hi=0, tile_dim0=4096
    g1[4] = 1;                          // tile_dim1=1, tile_dim2=0
    g1[5] = (int)elems;                 // tensor_dim0_stride lo32
    g1[6] = 0;                          // stride0 hi, stride1 lo
    g1[7] = 0;                          // stride1 hi
    i32x4 z4 = {};
#if __clang_major__ >= 23
    i32x8 z8 = {};
    __builtin_amdgcn_tensor_load_to_lds(g0, g1, z4, z4, z8, 0);
#else
    __builtin_amdgcn_tensor_load_to_lds(g0, g1, z4, z4, 0);
#endif
    __builtin_amdgcn_s_wait_tensorcnt(0);
  }
#else
  {
    const uint4* s = (const uint4*)g2wt; uint4* d = (uint4*)wl;
    for (int i = threadIdx.x; i < (128 * 128) / 8; i += 256) d[i] = s[i];
  }
#endif
  if (threadIdx.x < 128) { bsum[threadIdx.x] = 0.0f; bsq[threadIdx.x] = 0.0f; }
  __syncthreads();

  int wave = threadIdx.x >> 5, lane = threadIdx.x & 31;
  int rowbase = blockIdx.x * 128 + wave * 16;
  const _Float16* arow = h + (long)rowbase * 128;
  __builtin_prefetch(arow, 0, 0);

  v8f acc[8] = {};
#pragma unroll
  for (int ks = 0; ks < 4; ++ks) {
    v16h a = load_a_frag(arow, 128, ks * 32);
    v16h b[8];
#pragma unroll
    for (int t = 0; t < 8; ++t) b[t] = load_b_frag(wl, 128, t * 16, ks * 32);
#pragma unroll
    for (int t = 0; t < 8; ++t) acc[t] = wmma_f32(a, b[t], acc[t]);
  }

  int hh = lane >> 4;
  int gg[8];
#pragma unroll
  for (int r = 0; r < 8; ++r) gg[r] = batch[rowbase + hh * 8 + r];
#pragma unroll
  for (int t = 0; t < 8; ++t) {
    int n = t * 16 + (lane & 15);
    float bias = g2b[n];
#pragma unroll
    for (int r = 0; r < 8; ++r) {
      int row = rowbase + hh * 8 + r;
      float v = acc[t][r] + bias - corr2[gg[r] * 128 + n];
      y[(long)row * 128 + n] = v;
      atomicAdd(&bsum[n], v);          // ds_add_f32
      atomicAdd(&bsq[n], v * v);
    }
  }
  __syncthreads();
  if (threadIdx.x < 128) {
    atomicAdd(&bn_sum[threadIdx.x], bsum[threadIdx.x]);
    atomicAdd(&bn_sq[threadIdx.x], bsq[threadIdx.x]);
  }
}

__global__ void k_bnfin(const float* __restrict__ bn_sum, const float* __restrict__ bn_sq,
                        const float* __restrict__ gamma, const float* __restrict__ beta,
                        float* __restrict__ scale, float* __restrict__ shift) {
  int n = threadIdx.x;
  if (n >= 128) return;
  float mean = bn_sum[n] * (1.0f / (float)NN);
  float var  = bn_sq[n] * (1.0f / (float)NN) - mean * mean;
  float sc   = gamma[n] * rsqrtf(var + 1e-5f);
  scale[n] = sc;
  shift[n] = beta[n] - mean * sc;
}

// out = x + y*scale[n] + shift[n], float4-vectorized
__global__ void k_out(const float* __restrict__ x, const float* __restrict__ y,
                      const float* __restrict__ scale, const float* __restrict__ shift,
                      float* __restrict__ out) {
  int i = blockIdx.x * 256 + threadIdx.x;         // NN*32 float4 chunks
  if (i >= NN * 32) return;
  int n = (i & 31) * 4;
  float4 xv = ((const float4*)x)[i];
  float4 yv = ((const float4*)y)[i];
  float4 o;
  o.x = xv.x + yv.x * scale[n]     + shift[n];
  o.y = xv.y + yv.y * scale[n + 1] + shift[n + 1];
  o.z = xv.z + yv.z * scale[n + 2] + shift[n + 2];
  o.w = xv.w + yv.w * scale[n + 3] + shift[n + 3];
  ((float4*)out)[i] = o;
}

// ---------------------------------------------------------------------------
// Host launcher
// ---------------------------------------------------------------------------
extern "C" void kernel_launch(void* const* d_in, const int* in_sizes, int n_in,
                              void* d_out, int out_size, void* d_ws, size_t ws_size,
                              hipStream_t stream) {
  (void)in_sizes; (void)n_in; (void)out_size; (void)ws_size;
  const float* x     = (const float*)d_in[0];
  const int*   batch = (const int*)d_in[2];
  const float* W1    = (const float*)d_in[6];
  const float* b1    = (const float*)d_in[7];
  const float* W2    = (const float*)d_in[8];
  const float* b2    = (const float*)d_in[9];
  const float* G1W   = (const float*)d_in[10];
  const float* G1b   = (const float*)d_in[11];
  const float* L1W   = (const float*)d_in[12];
  const float* G2W   = (const float*)d_in[13];
  const float* G2b   = (const float*)d_in[14];
  const float* L2W   = (const float*)d_in[15];
  const float* gamma = (const float*)d_in[16];
  const float* beta  = (const float*)d_in[17];
  float* out = (float*)d_out;

  char* ws = (char*)d_ws;
  size_t cur = 0;
  auto alloc = [&](size_t bytes) -> void* {
    void* p = ws + cur;
    cur = (cur + bytes + 255) & ~(size_t)255;
    return p;
  };

  // zero-initialized accumulator region (first in ws)
  float* sum1   = (float*)alloc((size_t)GG * 160 * 4);
  float* cnt    = (float*)alloc((size_t)GG * 4);
  float* sum2   = (float*)alloc((size_t)GG * 128 * 4);
  float* bn_sum = (float*)alloc(128 * 4);
  float* bn_sq  = (float*)alloc(128 * 4);
  size_t zero_floats = cur / 4;

  float* corr1 = (float*)alloc((size_t)GG * 128 * 4);
  float* corr2 = (float*)alloc((size_t)GG * 128 * 4);
  float* scale = (float*)alloc(128 * 4);
  float* shift = (float*)alloc(128 * 4);
  _Float16* W1t  = (_Float16*)alloc(128 * 128 * 2);
  _Float16* W2t  = (_Float16*)alloc(16 * 128 * 2);
  _Float16* G1Wt = (_Float16*)alloc(128 * 160 * 2);
  _Float16* L1Wt = (_Float16*)alloc(128 * 160 * 2);
  _Float16* G2Wt = (_Float16*)alloc(128 * 128 * 2);
  _Float16* L2Wt = (_Float16*)alloc(128 * 128 * 2);
  _Float16* x0   = (_Float16*)alloc((size_t)NN * 160 * 2);
  _Float16* hbuf = (_Float16*)alloc((size_t)NN * 128 * 2);
  float*    ybuf = (float*)alloc((size_t)NN * 128 * 4);

  k_zero<<<(int)((zero_floats + 255) / 256), 256, 0, stream>>>((float*)ws, (int)zero_floats);

  k_prep_w<<<(128 * 128 + 255) / 256, 256, 0, stream>>>(W1, W1t, 128, 128, 128, 128);
  k_prep_w<<<(16 * 128 + 255) / 256, 256, 0, stream>>>(W2, W2t, 128, NFILT, 16, 128);
  k_prep_w<<<(128 * 160 + 255) / 256, 256, 0, stream>>>(G1W, G1Wt, 136, 128, 128, 160);
  k_prep_w<<<(128 * 160 + 255) / 256, 256, 0, stream>>>(L1W, L1Wt, 136, 128, 128, 160);
  k_prep_w<<<(128 * 128 + 255) / 256, 256, 0, stream>>>(G2W, G2Wt, 128, 128, 128, 128);
  k_prep_w<<<(128 * 128 + 255) / 256, 256, 0, stream>>>(L2W, L2Wt, 128, 128, 128, 128);

  k_convert_x<<<NN * 20 / 256, 256, 0, stream>>>(x, x0);
  k_filt<<<NN / 128, 256, 0, stream>>>(W1t, W2t, b1, b2, x0);
  k_scatter1<<<(NN * 136 + 255) / 256, 256, 0, stream>>>(x0, batch, sum1, cnt);
  k_meancorr<<<GG / 128, 256, 0, stream>>>(sum1, cnt, L1Wt, corr1, 160, 5);
  k_layer1<<<NN / 128, 256, 0, stream>>>(x0, G1Wt, G1b, corr1, batch, hbuf, sum2);
  k_meancorr<<<GG / 128, 256, 0, stream>>>(sum2, cnt, L2Wt, corr2, 128, 4);
  k_layer2<<<NN / 128, 256, 0, stream>>>(hbuf, G2Wt, G2b, corr2, batch, ybuf, bn_sum, bn_sq);
  k_bnfin<<<1, 128, 0, stream>>>(bn_sum, bn_sq, gamma, beta, scale, shift);
  k_out<<<NN * 32 / 256, 256, 0, stream>>>(x, ybuf, scale, shift, out);
}
